// GAT_Decoder_48155173322925
// MI455X (gfx1250) — matrix operand order repaired and checked by reading, not compiled, not run on hardware
//
#include <hip/hip_runtime.h>
#include <hip/hip_bf16.h>
#include <math.h>

// ---------------------------------------------------------------------------
// GAT decoder for MI455X (gfx1250, wave32, WMMA).
//   layer: h = T(x) @ W (wmma f16->f32), alpha_s/d = h . a  (from fragments)
//          edge softmax (3 memory passes, fp32 atomics in L2)
//          y = (agg + b) @ fW (wmma) + per-column BN stats (shfl reduce)
//          BN finalize -> scale/shift folded into next layer's A-load (+relu)
// ---------------------------------------------------------------------------

typedef __attribute__((ext_vector_type(16))) _Float16 v16h;
typedef __attribute__((ext_vector_type(8)))  float    v8f;

#define HD 64
#define NEG_SLOPE 0.2f
#define BN_EPS 1e-5f

__global__ void fill_kernel(float* __restrict__ p, size_t n, float v) {
  size_t i = (size_t)blockIdx.x * blockDim.x + threadIdx.x;
  if (i < n) p[i] = v;
}

__device__ inline void atomic_max_float(float* addr, float v) {
  // order-preserving int punning: works for mixed signs with -inf init
  if (v >= 0.f) atomicMax((int*)addr, __float_as_int(v));
  else          atomicMin((unsigned int*)addr, __float_as_uint(v));
}

// H[n,64] = T(X)[n,64] @ W[64,64]; T(x)[.,k] = relu?(x*scale[k]+shift[k])
// If avec_s: also alpha_s[n]=H[n,:].avec_s, alpha_d likewise (GAT mode).
// If stat_sum: also atomically accumulate per-column sum / sum-of-squares.
__global__ __launch_bounds__(256)
void gemm64_kernel(const float* __restrict__ X, const float* __restrict__ W,
                   const float* __restrict__ in_scale, const float* __restrict__ in_shift,
                   int relu_in,
                   const float* __restrict__ avec_s, const float* __restrict__ avec_d,
                   float* __restrict__ H,
                   float* __restrict__ alpha_s, float* __restrict__ alpha_d,
                   float* __restrict__ stat_sum, float* __restrict__ stat_sq,
                   int nrows)
{
  __shared__ __align__(32) _Float16 lds_wt[HD * HD];  // Wt[col][k], f16
  __shared__ float lds_s[HD], lds_t[HD], lds_vs[HD], lds_vd[HD];

  for (int i = threadIdx.x; i < HD * HD; i += 256) {
    int kk = i >> 6, cc = i & 63;              // W row-major [k][c]
    lds_wt[cc * HD + kk] = (_Float16)W[i];     // transpose into [c][k]
  }
  if (threadIdx.x < HD) {
    int c = threadIdx.x;
    lds_s[c]  = in_scale ? in_scale[c] : 1.f;
    lds_t[c]  = in_shift ? in_shift[c] : 0.f;
    lds_vs[c] = avec_s ? avec_s[c] : 0.f;
    lds_vd[c] = avec_d ? avec_d[c] : 0.f;
  }
  __syncthreads();

  const int wave = threadIdx.x >> 5;
  const int m0   = (blockIdx.x * 8 + wave) * 16;   // 16-row tile per wave
  if (m0 >= nrows) return;                          // nrows % 16 == 0

  const int lane = threadIdx.x & 31;
  const int half = lane >> 4;     // K-half selector (ISA 7.12.2 layouts)
  const int r    = lane & 15;     // row (A) / col (B,C) within tile

  const float* xrow = X + (size_t)(m0 + r) * HD;

  v8f acc0 = {}, acc1 = {}, acc2 = {}, acc3 = {};

#pragma unroll
  for (int ks = 0; ks < 64; ks += 32) {
    // A fragment 16x32 f16: lanes 0-15 hold K ks+0..15, lanes
    // 16-31 hold K ks+16..31 (row = m0 + r for all lanes)
    v16h a;
#pragma unroll
    for (int j = 0; j < 8; ++j) {
      int k0 = ks + half * 16 + j;
      int k1 = ks + half * 16 + 8 + j;
      float f0 = xrow[k0] * lds_s[k0] + lds_t[k0];
      float f1 = xrow[k1] * lds_s[k1] + lds_t[k1];
      if (relu_in) { f0 = fmaxf(f0, 0.f); f1 = fmaxf(f1, 0.f); }
      a[j]     = (_Float16)f0;
      a[8 + j] = (_Float16)f1;
    }
    // B fragment 32x16: lane holds one column, half selects K 0-15/16-31;
    // Wt[col][k] is contiguous in k -> one 32B LDS read per tile.
    v16h b0 = *(const v16h*)&lds_wt[( 0 + r) * HD + ks + half * 16];
    v16h b1 = *(const v16h*)&lds_wt[(16 + r) * HD + ks + half * 16];
    v16h b2 = *(const v16h*)&lds_wt[(32 + r) * HD + ks + half * 16];
    v16h b3 = *(const v16h*)&lds_wt[(48 + r) * HD + ks + half * 16];
    acc0 = __builtin_amdgcn_wmma_f32_16x16x32_f16(false, a, false, b0, (short)0, acc0, false, false);
    acc1 = __builtin_amdgcn_wmma_f32_16x16x32_f16(false, a, false, b1, (short)0, acc1, false, false);
    acc2 = __builtin_amdgcn_wmma_f32_16x16x32_f16(false, a, false, b2, (short)0, acc2, false, false);
    acc3 = __builtin_amdgcn_wmma_f32_16x16x32_f16(false, a, false, b3, (short)0, acc3, false, false);
  }

  // C/D layout: VGPR i -> row m0 + half*8 + i, lane r -> column
#pragma unroll
  for (int i = 0; i < 8; ++i) {
    float* hr = H + (size_t)(m0 + half * 8 + i) * HD;
    hr[ 0 + r] = acc0[i];
    hr[16 + r] = acc1[i];
    hr[32 + r] = acc2[i];
    hr[48 + r] = acc3[i];
  }

  if (alpha_s) {  // GAT mode: alpha = h . a, reduced over the 16 N-lanes
    float vs0 = lds_vs[r], vs1 = lds_vs[16 + r], vs2 = lds_vs[32 + r], vs3 = lds_vs[48 + r];
    float vd0 = lds_vd[r], vd1 = lds_vd[16 + r], vd2 = lds_vd[32 + r], vd3 = lds_vd[48 + r];
#pragma unroll
    for (int i = 0; i < 8; ++i) {
      float s = acc0[i] * vs0 + acc1[i] * vs1 + acc2[i] * vs2 + acc3[i] * vs3;
      float d = acc0[i] * vd0 + acc1[i] * vd1 + acc2[i] * vd2 + acc3[i] * vd3;
#pragma unroll
      for (int m = 1; m <= 8; m <<= 1) {
        s += __shfl_xor(s, m, 32);
        d += __shfl_xor(d, m, 32);
      }
      if (r == 0) {
        alpha_s[m0 + half * 8 + i] = s;
        alpha_d[m0 + half * 8 + i] = d;
      }
    }
  }

  if (stat_sum) {  // FC mode: per-column sum / sumsq for BatchNorm
    float s0 = 0, q0 = 0, s1 = 0, q1 = 0, s2 = 0, q2 = 0, s3 = 0, q3 = 0;
#pragma unroll
    for (int i = 0; i < 8; ++i) {
      float v;
      v = acc0[i]; s0 += v; q0 += v * v;
      v = acc1[i]; s1 += v; q1 += v * v;
      v = acc2[i]; s2 += v; q2 += v * v;
      v = acc3[i]; s3 += v; q3 += v * v;
    }
    s0 += __shfl_xor(s0, 16, 32); q0 += __shfl_xor(q0, 16, 32);
    s1 += __shfl_xor(s1, 16, 32); q1 += __shfl_xor(q1, 16, 32);
    s2 += __shfl_xor(s2, 16, 32); q2 += __shfl_xor(q2, 16, 32);
    s3 += __shfl_xor(s3, 16, 32); q3 += __shfl_xor(q3, 16, 32);
    if (half == 0) {
      atomicAdd(&stat_sum[ 0 + r], s0); atomicAdd(&stat_sq[ 0 + r], q0);
      atomicAdd(&stat_sum[16 + r], s1); atomicAdd(&stat_sq[16 + r], q1);
      atomicAdd(&stat_sum[32 + r], s2); atomicAdd(&stat_sq[32 + r], q2);
      atomicAdd(&stat_sum[48 + r], s3); atomicAdd(&stat_sq[48 + r], q3);
    }
  }
}

// pass 1: e = leakyrelu(as[src]+ad[dst]); m[dst] = max(m[dst], e)
__global__ void edge_max_kernel(const long long* __restrict__ ei, long long E, long long ET,
                                const float* __restrict__ as, const float* __restrict__ ad,
                                float* __restrict__ ebuf, float* __restrict__ mmax)
{
  long long t = (long long)blockIdx.x * blockDim.x + threadIdx.x;
  if (t >= ET) return;
  long long s, d;
  if (t < E) { s = ei[t]; d = ei[E + t]; } else { s = d = t - E; }
  float e = as[s] + ad[d];
  e = (e > 0.f) ? e : NEG_SLOPE * e;
  ebuf[t] = e;
  atomic_max_float(&mmax[d], e);
}

// pass 2: ex = exp(e - m[dst]); denom[dst] += ex
__global__ void edge_expsum_kernel(const long long* __restrict__ ei, long long E, long long ET,
                                   const float* __restrict__ mmax,
                                   float* __restrict__ ebuf, float* __restrict__ denom)
{
  long long t = (long long)blockIdx.x * blockDim.x + threadIdx.x;
  if (t >= ET) return;
  long long d = (t < E) ? ei[E + t] : t - E;
  float ex = __expf(ebuf[t] - mmax[d]);
  ebuf[t] = ex;
  atomicAdd(&denom[d], ex);
}

// pass 3: agg[dst] += (ex/denom[dst]) * h[src]; 16 threads/edge, float4 gather
__global__ void edge_agg_kernel(const long long* __restrict__ ei, long long E, long long ET,
                                const float* __restrict__ ebuf, const float* __restrict__ denom,
                                const float* __restrict__ Hs, float* __restrict__ agg)
{
  long long g = (long long)blockIdx.x * blockDim.x + threadIdx.x;
  long long t = g >> 4;
  if (t >= ET) return;
  int fq = (int)(g & 15) * 4;
  long long s, d;
  if (t < E) { s = ei[t]; d = ei[E + t]; } else { s = d = t - E; }
  float w = ebuf[t] / denom[d];
  float4 hv = *(const float4*)(Hs + s * HD + fq);
  float* ap = agg + d * HD + fq;
  atomicAdd(ap + 0, w * hv.x);
  atomicAdd(ap + 1, w * hv.y);
  atomicAdd(ap + 2, w * hv.z);
  atomicAdd(ap + 3, w * hv.w);
}

__global__ void bn_finalize_kernel(const float* __restrict__ sum, const float* __restrict__ sq,
                                   const float* __restrict__ g, const float* __restrict__ be,
                                   float* __restrict__ scale, float* __restrict__ shift, float invN)
{
  int c = threadIdx.x;
  if (c >= HD) return;
  float mu  = sum[c] * invN;
  float var = sq[c] * invN - mu * mu;  // biased, as torch BN training fwd
  float sc  = g[c] * rsqrtf(var + BN_EPS);
  scale[c] = sc;
  shift[c] = be[c] - mu * sc;
}

__global__ void bn_relu_out_kernel(const float* __restrict__ y,
                                   const float* __restrict__ scale, const float* __restrict__ shift,
                                   float* __restrict__ out, size_t n)
{
  size_t i = (size_t)blockIdx.x * blockDim.x + threadIdx.x;
  if (i >= n) return;
  int c = (int)(i & 63);
  out[i] = fmaxf(y[i] * scale[c] + shift[c], 0.f);
}

extern "C" void kernel_launch(void* const* d_in, const int* in_sizes, int n_in,
                              void* d_out, int out_size, void* d_ws, size_t ws_size,
                              hipStream_t stream)
{
  const float*     x   = (const float*)d_in[0];
  const long long* ei  = (const long long*)d_in[1];   // int64 [2, E]
  const float* W1  = (const float*)d_in[2];
  const float* as1 = (const float*)d_in[3];
  const float* ad1 = (const float*)d_in[4];
  const float* b1  = (const float*)d_in[5];
  const float* fW1 = (const float*)d_in[6];
  // d_in[7] = fb1: cancels inside BatchNorm -> unused
  const float* g1  = (const float*)d_in[8];
  const float* be1 = (const float*)d_in[9];
  const float* W2  = (const float*)d_in[10];
  const float* as2 = (const float*)d_in[11];
  const float* ad2 = (const float*)d_in[12];
  const float* b2  = (const float*)d_in[13];
  const float* fW2 = (const float*)d_in[14];
  // d_in[15] = fb2: cancels inside BatchNorm -> unused
  const float* g2  = (const float*)d_in[16];
  const float* be2 = (const float*)d_in[17];
  float* out = (float*)d_out;

  const int       N  = in_sizes[0] / HD;      // 100000 (multiple of 16)
  const long long E  = in_sizes[1] / 2;       // 1600000
  const long long ET = E + N;                 // with self-loops

  // workspace layout (floats), ~86 MB total
  float* p = (float*)d_ws;
  float* buf_h   = p; p += (size_t)N * HD;
  float* buf_agg = p; p += (size_t)N * HD;
  float* buf_y   = p; p += (size_t)N * HD;
  float* buf_as  = p; p += N;
  float* buf_ad  = p; p += N;
  float* buf_m   = p; p += N;
  float* buf_den = p; p += N;
  float* buf_e   = p; p += ET;
  float* st_sum  = p; p += HD;   // st_sum/st_sq contiguous (one 128-elt fill)
  float* st_sq   = p; p += HD;
  float* bn_sc   = p; p += HD;
  float* bn_sh   = p; p += HD;

  auto cdiv = [](long long a, long long b) { return (int)((a + b - 1) / b); };
  const dim3 b256(256);
  const int gemm_blocks = cdiv((long long)N / 16, 8);   // 1 wave = 16 rows, 8 waves/block
  const float invN = 1.f / (float)N;

  for (int layer = 0; layer < 2; ++layer) {
    const float* Wg  = layer ? W2  : W1;
    const float* avs = layer ? as2 : as1;
    const float* avd = layer ? ad2 : ad1;
    const float* bg  = layer ? b2  : b1;
    const float* fW  = layer ? fW2 : fW1;
    const float* gg  = layer ? g2  : g1;
    const float* be  = layer ? be2 : be1;
    const float* Xin = layer ? buf_y : x;
    const float* sc  = layer ? bn_sc : nullptr;   // fold BN+ReLU of layer 1
    const float* sh  = layer ? bn_sh : nullptr;   // into layer 2's A-load
    const int relu_in = layer ? 1 : 0;

    fill_kernel<<<cdiv(N, 256), b256, 0, stream>>>(buf_m, (size_t)N, -INFINITY);
    fill_kernel<<<cdiv(N, 256), b256, 0, stream>>>(buf_den, (size_t)N, 0.f);
    fill_kernel<<<cdiv((long long)N * HD, 256), b256, 0, stream>>>(buf_agg, (size_t)N * HD, 0.f);
    fill_kernel<<<1, b256, 0, stream>>>(st_sum, (size_t)(2 * HD), 0.f);

    // GAT: h = T(Xin) @ W  (+ alpha_s/alpha_d from fragments)
    gemm64_kernel<<<gemm_blocks, b256, 0, stream>>>(
        Xin, Wg, sc, sh, relu_in, avs, avd,
        buf_h, buf_as, buf_ad, nullptr, nullptr, N);

    // edge softmax + weighted scatter
    edge_max_kernel   <<<cdiv(ET, 256),      b256, 0, stream>>>(ei, E, ET, buf_as, buf_ad, buf_e, buf_m);
    edge_expsum_kernel<<<cdiv(ET, 256),      b256, 0, stream>>>(ei, E, ET, buf_m, buf_e, buf_den);
    edge_agg_kernel   <<<cdiv(ET * 16, 256), b256, 0, stream>>>(ei, E, ET, buf_e, buf_den, buf_h, buf_agg);

    // FC: y = (agg + b) @ fW  (+ per-column BN stats; fb cancels in BN)
    gemm64_kernel<<<gemm_blocks, b256, 0, stream>>>(
        buf_agg, fW, nullptr, bg, 0, nullptr, nullptr,
        buf_y, nullptr, nullptr, st_sum, st_sq, N);

    bn_finalize_kernel<<<1, 64, 0, stream>>>(st_sum, st_sq, gg, be, bn_sc, bn_sh, invN);
  }

  // final BN+ReLU of layer 2 -> output
  bn_relu_out_kernel<<<cdiv((long long)N * HD, 256), b256, 0, stream>>>(
      buf_y, bn_sc, bn_sh, out, (size_t)N * HD);
}
